// style_bottleneck_IR_27367531610881
// MI455X (gfx1250) — matrix operand order repaired and checked by reading, not compiled, not run on hardware
//
#include <hip/hip_runtime.h>

typedef __attribute__((ext_vector_type(16))) __bf16 v16bf;
typedef __attribute__((ext_vector_type(8)))  __bf16 v8bf;
typedef __attribute__((ext_vector_type(8)))  float  v8f;
typedef __attribute__((ext_vector_type(4)))  unsigned int v4u;
typedef __attribute__((ext_vector_type(8)))  int v8i;
typedef __attribute__((ext_vector_type(4)))  int v4i;

#define CC      64
#define HH      256
#define WW      256
#define BBATCH  8
#define PW      258            // padded H/W for NHWC intermediates
#define KDIM    576            // 9 taps * 64 ci,  k = tap*64 + ci
#define TILE_N  128
#define LIN_SCALE  0.04419417382415922f   // 1/sqrt(512)
#define CONV_SCALE 0.041666666666666664f  // 1/sqrt(64*9)
#define FUSED_SCALE 1.4142135623730951f

__device__ __forceinline__ int nhwc_idx(int b, int yp, int xp, int c) {
    return ((b * PW + yp) * PW + xp) * CC + c;
}

// ---- TDM: 2-D tile load (data_size = 8B units), D# per CDNA5 ISA ch.8 ----
__device__ __forceinline__ void tdm_load_2d(unsigned int lds_byte_addr,
                                            const void* gptr,
                                            unsigned int tile_d0,          // 8B elems per row
                                            unsigned int tile_d1,          // rows
                                            unsigned long long stride_d0)  // 8B elems
{
    unsigned long long ga = (unsigned long long)(uintptr_t)gptr;
    v4u g0;
    g0[0] = 1u;                                             // count=1, user mode
    g0[1] = lds_byte_addr;                                  // lds_addr
    g0[2] = (unsigned int)ga;                               // global_addr[31:0]
    g0[3] = (unsigned int)((ga >> 32) & 0x1FFFFFFull)       // global_addr[56:32]
            | (2u << 30);                                   // type=2 (image)
    v8i g1;
    g1[0] = (int)(3u << 16);                                // data_size=8B, no mask
    g1[1] = (int)((tile_d0 & 0xFFFFu) << 16);               // tensor_dim0[15:0]
    g1[2] = (int)(((tile_d0 >> 16) & 0xFFFFu)               // tensor_dim0[31:16]
            | ((tile_d1 & 0xFFFFu) << 16));                 // tensor_dim1[15:0]
    g1[3] = (int)((tile_d0 & 0xFFFFu) << 16);               // tensor_dim1 hi=0 | tile_dim0
    g1[4] = (int)(tile_d1 & 0xFFFFu);                       // tile_dim1 (tile_dim2=0)
    g1[5] = (int)(unsigned int)(stride_d0 & 0xFFFFFFFFull); // dim0_stride[31:0]
    g1[6] = (int)(unsigned int)((stride_d0 >> 32) & 0xFFFFull); // dim0_stride[47:32]
    g1[7] = 0;
    v4i z4 = {0, 0, 0, 0};
    v8i z8 = {0, 0, 0, 0, 0, 0, 0, 0};
    __builtin_amdgcn_tensor_load_to_lds(g0, g1, z4, z4, z8, 0);
}

// ---------------------------------------------------------------- prep kernels

__global__ void style_mod_kernel(const float* __restrict__ style,
                                 const float* __restrict__ mod_w,
                                 const float* __restrict__ mod_b,
                                 float* __restrict__ s_out) {
    int t = threadIdx.x;            // 512 threads: b = t/64, ci = t%64
    int b = t >> 6, ci = t & 63;
    float acc = 0.f;
    for (int k = 0; k < 512; ++k)
        acc += style[b * 512 + k] * mod_w[ci * 512 + k];
    s_out[t] = acc * LIN_SCALE + mod_b[ci];
}

__global__ void reorder_w_kernel(const float* __restrict__ w,
                                 __bf16* __restrict__ dst) {
    int co = blockIdx.x, ci = threadIdx.x;
    for (int t = 0; t < 9; ++t)
        dst[co * KDIM + t * 64 + ci] = (__bf16)(w[(co * 64 + ci) * 9 + t]);
}

__global__ void mod_w_kernel(const float* __restrict__ conv_w,
                             const float* __restrict__ s,
                             __bf16* __restrict__ dst) {
    int co = blockIdx.x, b = blockIdx.y, ci = threadIdx.x;
    float sv = s[b * 64 + ci];
    float wv[9];
    float sq = 0.f;
    for (int t = 0; t < 9; ++t) {
        float v = CONV_SCALE * conv_w[(co * 64 + ci) * 9 + t] * sv;
        wv[t] = v;
        sq += v * v;
    }
    __shared__ float red[64];
    red[ci] = sq;
    __syncthreads();
    for (int off = 32; off > 0; off >>= 1) {
        if (ci < off) red[ci] += red[ci + off];
        __syncthreads();
    }
    float demod = rsqrtf(red[0] + 1e-8f);
    for (int t = 0; t < 9; ++t)
        dst[(b * 64 + co) * KDIM + t * 64 + ci] = (__bf16)(wv[t] * demod);
}

// zero the 1-pixel borders of both padded NHWC ping-pong buffers
__global__ void border_zero_kernel(__bf16* __restrict__ a,
                                   __bf16* __restrict__ c) {
    int i = blockIdx.x * blockDim.x + threadIdx.x;   // over B*PW*PW pixel slots
    if (i >= BBATCH * PW * PW) return;
    int xp = i % PW;
    int yp = (i / PW) % PW;
    if (xp == 0 || xp == PW - 1 || yp == 0 || yp == PW - 1) {
        uint4 z = {0u, 0u, 0u, 0u};
        uint4* pa = (uint4*)&a[(size_t)i * CC];
        uint4* pc = (uint4*)&c[(size_t)i * CC];
#pragma unroll
        for (int j = 0; j < 8; ++j) { pa[j] = z; pc[j] = z; }
    }
}

// ---------------------------------------------------------------- main conv
// MODE 0: out = prelu(conv(in,w), aux=alpha[64])
// MODE 1: out = aux(residual tensor) + conv(in,w)
// MODE 2: out = leaky(conv(in,w_b)+aux=bias, 0.2)*sqrt2, per-batch weights
// IN32 : input is f32 NCHW (else bf16 padded NHWC, staged by TDM)
// RES32: MODE1 residual is f32 NCHW (else bf16 padded NHWC)
// OUT32: output is f32 NCHW (else bf16 padded NHWC)
template <int MODE, bool IN32, bool RES32, bool OUT32>
__global__ __launch_bounds__(256)
void conv3x3_wmma_kernel(const void* __restrict__ in_,
                         const __bf16* __restrict__ wq,
                         const void* __restrict__ aux_,
                         void* __restrict__ out_) {
    __shared__ __align__(32) __bf16 w_lds[CC * KDIM];      // 73728 B
    __shared__ __align__(32) __bf16 in_lds[3 * 130 * CC];  // 49920 B

    const int tid = threadIdx.x;
    const int b   = blockIdx.z;
    const int y   = blockIdx.y;
    const int x0  = blockIdx.x * TILE_N;
    const int wv  = tid >> 5;

    // ---- async staging via Tensor Data Mover (wave 0 issues) -------------
    if (wv == 0) {
        const __bf16* wsrc = wq + (MODE == 2 ? b * (CC * KDIM) : 0);
        tdm_load_2d((unsigned int)(uintptr_t)w_lds, wsrc,
                    (CC * KDIM * 2) / 8, 1, (CC * KDIM * 2) / 8);
        if constexpr (!IN32) {
            const __bf16* src = (const __bf16*)in_ +
                (size_t)nhwc_idx(b, y, x0, 0);   // rows y..y+2 = halo of out row y
            tdm_load_2d((unsigned int)(uintptr_t)in_lds, src,
                        (130 * CC * 2) / 8, 3, ((unsigned long long)PW * CC * 2) / 8);
        }
        __builtin_amdgcn_s_wait_tensorcnt(0);
    }
    if constexpr (IN32) {   // f32 NCHW input: convert + transpose into LDS
        const float* in = (const float*)in_;
        for (int i = tid; i < 3 * 130 * CC; i += 256) {
            int x   = i % 130;
            int row = (i / 130) % 3;
            int ci  = i / 390;
            int gx = x0 + x - 1;
            int gy = y + row - 1;
            float v = 0.f;
            if ((unsigned)gx < (unsigned)WW && (unsigned)gy < (unsigned)HH)
                v = in[((b * CC + ci) * HH + gy) * WW + gx];
            in_lds[(row * 130 + x) * CC + ci] = (__bf16)v;
        }
    }
    __syncthreads();

    // ---- implicit GEMM: M=64(co), N=128(px), K=576; 2Mx2N tiles per wave --
    const int lane = tid & 31;
    const int lg   = lane >> 4;
    const int idx  = lane & 15;
    const int m0t  = (wv & 1) * 2;   // two M tiles: m0t, m0t+1
    const int n0t  = (wv >> 1) * 2;  // two N tiles: n0t, n0t+1

    v8f acc[2][2] = {};

    const __bf16* wrow0 = &w_lds[((m0t + 0) * 16 + idx) * KDIM];
    const __bf16* wrow1 = &w_lds[((m0t + 1) * 16 + idx) * KDIM];
    for (int kc = 0; kc < 18; ++kc) {
        const int k0a = kc * 32;
        // A frags: lanes 0-15 hold K{0-15}, lanes 16-31 K{16-31}
        union { v16bf v; v8bf h[2]; } A0, A1;
        A0.v = *(const v16bf*)&wrow0[k0a + lg * 16];
        A1.v = *(const v16bf*)&wrow1[k0a + lg * 16];
        // B frags: lane = column N, K contiguous 16
        const int k0b = k0a + lg * 16;
        const int t   = k0b >> 6;
        const int ci0 = k0b & 63;
        const int ky  = t / 3;
        const int kx  = t - ky * 3;
        const __bf16* brow = &in_lds[(ky * 130 + kx) * CC + ci0];
        v16bf B0 = *(const v16bf*)&brow[((n0t + 0) * 16 + idx) * CC];
        v16bf B1 = *(const v16bf*)&brow[((n0t + 1) * 16 + idx) * CC];
        acc[0][0] = __builtin_amdgcn_wmma_f32_16x16x32_bf16(false, A0.v, false, B0, (short)0, acc[0][0], false, false);
        acc[0][1] = __builtin_amdgcn_wmma_f32_16x16x32_bf16(false, A0.v, false, B1, (short)0, acc[0][1], false, false);
        acc[1][0] = __builtin_amdgcn_wmma_f32_16x16x32_bf16(false, A1.v, false, B0, (short)0, acc[1][0], false, false);
        acc[1][1] = __builtin_amdgcn_wmma_f32_16x16x32_bf16(false, A1.v, false, B1, (short)0, acc[1][1], false, false);
    }

    // ---- fused epilogue + store (branchless activations) -----------------
#pragma unroll
    for (int mi = 0; mi < 2; ++mi) {
        const int cobase = (m0t + mi) * 16 + lg * 8;   // 8 contiguous co per lane
#pragma unroll
        for (int ni = 0; ni < 2; ++ni) {
            const int p = (n0t + ni) * 16 + idx;       // pixel within tile
            float f[8];
#pragma unroll
            for (int r = 0; r < 8; ++r) f[r] = acc[mi][ni][r];

            if constexpr (MODE == 0) {
                const float* alpha = (const float*)aux_;
#pragma unroll
                for (int r = 0; r < 8; ++r) {
                    float al = alpha[cobase + r];
                    // prelu(v) = max(v,0) + alpha*min(v,0)   (branchless)
                    f[r] = fmaxf(f[r], 0.f) + al * fminf(f[r], 0.f);
                }
            } else if constexpr (MODE == 1) {
                if constexpr (RES32) {
                    const float* res = (const float*)aux_;
#pragma unroll
                    for (int r = 0; r < 8; ++r)
                        f[r] += res[((b * CC + cobase + r) * HH + y) * WW + x0 + p];
                } else {
                    const __bf16* res = (const __bf16*)aux_;
                    union { uint4 u; __bf16 h[8]; } R;
                    R.u = *(const uint4*)&res[nhwc_idx(b, y + 1, x0 + p + 1, cobase)];
#pragma unroll
                    for (int r = 0; r < 8; ++r) f[r] += (float)R.h[r];
                }
            } else {
                const float* bias = (const float*)aux_;
#pragma unroll
                for (int r = 0; r < 8; ++r) {
                    float v = f[r] + bias[cobase + r];
                    // leaky(v,0.2)*sqrt2 = sqrt2*max(v,0) + 0.2*sqrt2*min(v,0)
                    f[r] = FUSED_SCALE * fmaxf(v, 0.f)
                         + (0.2f * FUSED_SCALE) * fminf(v, 0.f);
                }
            }

            if constexpr (OUT32) {
                float* out = (float*)out_;
#pragma unroll
                for (int r = 0; r < 8; ++r)
                    out[((b * CC + cobase + r) * HH + y) * WW + x0 + p] = f[r];
            } else {
                __bf16* out = (__bf16*)out_;
                union { uint4 u; __bf16 h[8]; } P;
#pragma unroll
                for (int r = 0; r < 8; ++r) P.h[r] = (__bf16)f[r];
                *(uint4*)&out[nhwc_idx(b, y + 1, x0 + p + 1, cobase)] = P.u;
            }
        }
    }
}

// ---------------------------------------------------------------- launcher

extern "C" void kernel_launch(void* const* d_in, const int* in_sizes, int n_in,
                              void* d_out, int out_size, void* d_ws, size_t ws_size,
                              hipStream_t stream) {
    const float* x       = (const float*)d_in[0];
    const float* style   = (const float*)d_in[1];
    const float* w1a     = (const float*)d_in[2];
    const float* alpha1  = (const float*)d_in[3];
    const float* w1b     = (const float*)d_in[4];
    const float* w2a     = (const float*)d_in[5];
    const float* alpha2  = (const float*)d_in[6];
    const float* w2b     = (const float*)d_in[7];
    const float* mod_w   = (const float*)d_in[8];
    const float* mod_b   = (const float*)d_in[9];
    const float* conv_w  = (const float*)d_in[10];
    const float* fused_b = (const float*)d_in[11];
    float* out = (float*)d_out;

    char* ws = (char*)d_ws;
    const size_t PB = (size_t)BBATCH * PW * PW * CC * sizeof(__bf16); // 68161536
    __bf16* bufA = (__bf16*)ws;
    __bf16* bufB = (__bf16*)(ws + PB);
    float*  sbuf = (float*)(ws + 2 * PB);
    __bf16* wq   = (__bf16*)(ws + 2 * PB + 4096);
    __bf16* wq1a = wq;
    __bf16* wq1b = wq + 1 * CC * KDIM;
    __bf16* wq2a = wq + 2 * CC * KDIM;
    __bf16* wq2b = wq + 3 * CC * KDIM;
    __bf16* wqm  = wq + 4 * CC * KDIM;   // 8 per-batch weight sets

    style_mod_kernel<<<1, 512, 0, stream>>>(style, mod_w, mod_b, sbuf);
    reorder_w_kernel<<<64, 64, 0, stream>>>(w1a, wq1a);
    reorder_w_kernel<<<64, 64, 0, stream>>>(w1b, wq1b);
    reorder_w_kernel<<<64, 64, 0, stream>>>(w2a, wq2a);
    reorder_w_kernel<<<64, 64, 0, stream>>>(w2b, wq2b);
    mod_w_kernel<<<dim3(64, 8), 64, 0, stream>>>(conv_w, sbuf, wqm);
    border_zero_kernel<<<(BBATCH * PW * PW + 255) / 256, 256, 0, stream>>>(bufA, bufB);

    dim3 grid(WW / TILE_N, HH, BBATCH), blk(256);
    // block1: t0 = prelu(conv(x,w1a)); t1 = x + conv(t0,w1b)
    conv3x3_wmma_kernel<0, true,  false, false><<<grid, blk, 0, stream>>>(x,    wq1a, alpha1,  bufA);
    conv3x3_wmma_kernel<1, false, true,  false><<<grid, blk, 0, stream>>>(bufA, wq1b, x,       bufB);
    // block2: t2 = prelu(conv(t1,w2a)); t3 = t1 + conv(t2,w2b) (in-place residual: safe)
    conv3x3_wmma_kernel<0, false, false, false><<<grid, blk, 0, stream>>>(bufB, wq2a, alpha2,  bufA);
    conv3x3_wmma_kernel<1, false, false, false><<<grid, blk, 0, stream>>>(bufA, wq2b, bufB,    bufB);
    // modulated conv + FusedLeakyReLU -> f32 NCHW d_out
    conv3x3_wmma_kernel<2, false, false, true ><<<grid, blk, 0, stream>>>(bufB, wqm,  fused_b, out);
}
